// GridRenderer_26869315404533
// MI455X (gfx1250) — compile-verified, matrix operand order checked by reading, not run on hardware
//
#include <hip/hip_runtime.h>

typedef _Float16 v16h __attribute__((ext_vector_type(16)));
typedef _Float16 v8h  __attribute__((ext_vector_type(8)));
typedef float    v8f  __attribute__((ext_vector_type(8)));

#define NLEV 16
#define TAB  (1 << 19)
#define AS   104            // activation row stride (halfs), padded vs 96 to dodge bank conflicts

// ---- LDS layout (units: halfs) ----
#define OFF_W0   0          // ws0 : KC=1 NT=4 -> 2048
#define OFF_W1   2048       // ws1 : KC=2 NT=4 -> 4096
#define OFF_W2   6144       // ws2 : KC=2 NT=5 -> 5120 (65 cols padded to 80)
#define OFF_WC0  11264      // wc0 : KC=3 NT=4 -> 6144 (K=80 padded to 96)
#define OFF_WC1  17408      // wc1 : KC=2 NT=1 -> 1024 (3 cols padded to 16)
#define OFF_ACT  18432      // 4 waves * 2 * 16 * AS = 13312
#define SMEM_HALFS 31744    // 63488 bytes -> 5 blocks (20 waves) per 320KB WGP

__constant__ float LSCALE[NLEV] = {
  15.0f, 19.158736798317971f, 24.398416831491192f, 31.0f,
  39.317473596635943f, 49.796833662982383f, 63.0f, 79.634947193271885f,
  100.59366732596479f, 127.0f, 160.26989438657756f, 202.18733465192957f,
  255.0f, 321.53978877315512f, 405.37466930385914f, 511.0f };
__constant__ int LRES[NLEV] = {
  16, 21, 26, 32, 41, 51, 64, 81, 102, 128, 162, 204, 256, 323, 407, 512 };

__device__ __forceinline__ v8f wmma16(v16h a, v16h b, v8f c) {
  // v_wmma_f32_16x16x32_f16 : D = A(16x32 f16) * B(32x16 f16) + C(16x16 f32)
  return __builtin_amdgcn_wmma_f32_16x16x32_f16(false, a, false, b, (short)0, c, false, false);
}

// Convert-then-relu in packed halves: 4x v_cvt_pk_f16_f32 + 4x v_pk_max_num_f16
// instead of 8x v_max_num_f32 + 8x v_cvt_f16_f32. relu commutes with the
// monotone f32->f16 conversion.
__device__ __forceinline__ v8h relu_h8(v8f acc) {
  v8h hv;
  #pragma unroll
  for (int r = 0; r < 8; ++r) hv[r] = (_Float16)acc[r];
  const v8h HZ = {(_Float16)0, (_Float16)0, (_Float16)0, (_Float16)0,
                  (_Float16)0, (_Float16)0, (_Float16)0, (_Float16)0};
  return __builtin_elementwise_max(hv, HZ);
}

// Cooperative: f32 weights -> f16, stored in LDS pre-swizzled in per-lane B-fragment
// order (lane = column n&15; K halves split by lane>>4), so a B fragment is one
// contiguous 32-byte LDS read per lane.
__device__ __forceinline__ void fill_w(const float* __restrict__ W, int realK, int realN,
                                       int KC, int NT, _Float16* dst, int tid) {
  const int total = KC * NT * 512;          // 32 lanes * 16 halfs per fragment
  for (int idx = tid; idx < total; idx += 128) {
    const int i    = idx & 15;
    const int lane = (idx >> 4) & 31;
    const int t    = idx >> 9;
    const int nt   = t % NT;
    const int kc   = t / NT;
    const int kh   = (lane >> 4) << 3;                       // 0 or 8
    const int k    = kc * 32 + ((i < 8) ? (kh + i) : (kh + i + 8));
    const int n    = nt * 16 + (lane & 15);
    const float v  = (k < realK && n < realN) ? W[k * realN + n] : 0.0f;
    dst[idx] = (_Float16)v;                                  // idx == fragment-major index
  }
}

// A fragment (16x32 f16, M row per lane) from row-major [16][AS] activation buffer:
// lane reads K runs [kh..kh+7] and [kh+16..kh+23] -> two ds_load_b128.
__device__ __forceinline__ v16h load_a(const _Float16* act, int lane, int kc) {
  const int mm = lane & 15;
  const int kh = kc * 32 + ((lane >> 4) << 3);
  const _Float16* p = act + mm * AS + kh;
  v8h lo = *(const v8h*)p;
  v8h hv = *(const v8h*)(p + 16);
  return __builtin_shufflevector(lo, hv, 0,1,2,3,4,5,6,7,8,9,10,11,12,13,14,15);
}

__device__ __forceinline__ v16h load_b(const _Float16* w, int NT, int kc, int nt, int lane) {
  return *(const v16h*)(w + ((((kc * NT + nt) << 5) + lane) << 4));
}

__global__ __launch_bounds__(128)
void ngp_fused_kernel(const float* __restrict__ x, const float* __restrict__ dvec,
                      const float* __restrict__ table,
                      const float* __restrict__ ws0, const float* __restrict__ ws1,
                      const float* __restrict__ ws2, const float* __restrict__ wc0,
                      const float* __restrict__ wc1,
                      float* __restrict__ out, int npts, int nb) {
  __shared__ _Float16 sm[SMEM_HALFS];
  const int tid = threadIdx.x;

  fill_w(ws0, 32, 64, 1, 4, sm + OFF_W0,  tid);
  fill_w(ws1, 64, 64, 2, 4, sm + OFF_W1,  tid);
  fill_w(ws2, 64, 65, 2, 5, sm + OFF_W2,  tid);
  fill_w(wc0, 80, 64, 3, 4, sm + OFF_WC0, tid);
  fill_w(wc1, 64,  3, 2, 1, sm + OFF_WC1, tid);
  __syncthreads();

  const int lane = tid & 31;
  const int wave = tid >> 5;
  const int m    = lane & 15;       // point row / tile column, depending on role
  const int hi   = lane >> 4;       // half-wave id
  const int mrb  = hi << 3;         // D-tile row base for this half-wave
  _Float16* actA = sm + OFF_ACT + wave * (2 * 16 * AS);
  _Float16* actC = actA + 16 * AS;
  const v8f ZACC = {0.f,0.f,0.f,0.f,0.f,0.f,0.f,0.f};

  for (int b = blockIdx.x; b < nb; b += gridDim.x) {
    const int pbase = b * 64 + wave * 16;
    const int p  = pbase + m;
    const int pc = (p < npts) ? p : (npts - 1);

    // ---------- hashgrid encode: lane (m, hi) handles point m, levels hi*8..hi*8+7 ----------
    {
      // x/d are streamed exactly once -> non-temporal so they never evict the
      // L2-resident 67MB hash table.
      const float gx = (__builtin_nontemporal_load(x + 3 * pc + 0) + 1.f) * 0.5f;
      const float gy = (__builtin_nontemporal_load(x + 3 * pc + 1) + 1.f) * 0.5f;
      const float gz = (__builtin_nontemporal_load(x + 3 * pc + 2) + 1.f) * 0.5f;
      #pragma unroll
      for (int li = 0; li < 8; ++li) {
        const int   l   = (hi << 3) + li;
        const float s   = LSCALE[l];
        const int   res = LRES[l];
        const float px = gx * s + 0.5f, py = gy * s + 0.5f, pz = gz * s + 0.5f;
        const float fpx = floorf(px), fpy = floorf(py), fpz = floorf(pz);
        const float frx = px - fpx, fry = py - fpy, frz = pz - fpz;
        const int ix = (int)fpx, iy = (int)fpy, iz = (int)fpz;
        const bool dense = (res * res * res) <= TAB;
        float f0 = 0.f, f1 = 0.f;
        #pragma unroll
        for (int c = 0; c < 8; ++c) {
          const int bx = c & 1, by = (c >> 1) & 1, bz = (c >> 2) & 1;
          const int cx = min(max(ix + bx, 0), res - 1);
          const int cy = min(max(iy + by, 0), res - 1);
          const int cz = min(max(iz + bz, 0), res - 1);
          unsigned idx;
          if (dense) idx = (unsigned)(cx + cy * res + cz * res * res);
          else       idx = ((unsigned)cx ^ ((unsigned)cy * 2654435761u)
                                         ^ ((unsigned)cz * 805459861u)) & (unsigned)(TAB - 1);
          const float w = (bx ? frx : 1.f - frx) * (by ? fry : 1.f - fry) * (bz ? frz : 1.f - frz);
          const float* tp = table + ((((size_t)l << 19) + idx) << 1);
          f0 += w * tp[0];
          f1 += w * tp[1];
        }
        actA[m * AS + 2 * l + 0] = (_Float16)f0;
        actA[m * AS + 2 * l + 1] = (_Float16)f1;
      }
    }

    // ---------- SH degree-4 (lanes 0..15) / zero K-pad of color input (lanes 16..31) ----------
    if (hi == 0) {
      const float dx = __builtin_nontemporal_load(dvec + 3 * pc + 0);
      const float dy = __builtin_nontemporal_load(dvec + 3 * pc + 1);
      const float dz = __builtin_nontemporal_load(dvec + 3 * pc + 2);
      const float xx = dx * dx, yy = dy * dy, zz = dz * dz;
      const float xy = dx * dy, yz = dy * dz, xz = dx * dz;
      _Float16* o = actC + m * AS;
      o[0]  = (_Float16)(0.28209479177387814f);
      o[1]  = (_Float16)(-0.48860251190291987f * dy);
      o[2]  = (_Float16)( 0.48860251190291987f * dz);
      o[3]  = (_Float16)(-0.48860251190291987f * dx);
      o[4]  = (_Float16)( 1.0925484305920792f * xy);
      o[5]  = (_Float16)(-1.0925484305920792f * yz);
      o[6]  = (_Float16)( 0.94617469575756f * zz - 0.31539156525252005f);
      o[7]  = (_Float16)(-1.0925484305920792f * xz);
      o[8]  = (_Float16)( 0.5462742152960396f * (xx - yy));
      o[9]  = (_Float16)( 0.5900435899266435f * dy * (-3.0f * xx + yy));
      o[10] = (_Float16)( 2.890611442640554f * xy * dz);
      o[11] = (_Float16)( 0.4570457994644657f * dy * (1.0f - 5.0f * zz));
      o[12] = (_Float16)( 0.3731763325901154f * dz * (5.0f * zz - 3.0f));
      o[13] = (_Float16)( 0.4570457994644657f * dx * (1.0f - 5.0f * zz));
      o[14] = (_Float16)( 1.445305721320277f * dz * (xx - yy));
      o[15] = (_Float16)( 0.5900435899266435f * dx * (-xx + 3.0f * yy));
    } else {
      #pragma unroll
      for (int j = 0; j < 16; ++j) actC[m * AS + 80 + j] = (_Float16)0.f;
    }

    // ---------- sigma MLP layer 0: 32 -> 64, relu (all 4 B frags preloaded) ----------
    {
      v16h a0 = load_a(actA, lane, 0);
      v16h b0 = load_b(sm + OFF_W0, 4, 0, 0, lane);
      v16h b1 = load_b(sm + OFF_W0, 4, 0, 1, lane);
      v16h b2 = load_b(sm + OFF_W0, 4, 0, 2, lane);
      v16h b3 = load_b(sm + OFF_W0, 4, 0, 3, lane);
      v16h bf[4] = {b0, b1, b2, b3};
      #pragma unroll
      for (int nt = 0; nt < 4; ++nt) {
        v8f acc = ZACC;
        acc = wmma16(a0, bf[nt], acc);
        v8h hv = relu_h8(acc);
        #pragma unroll
        for (int r = 0; r < 8; ++r)
          actA[(mrb + r) * AS + nt * 16 + m] = hv[r];
      }
    }
    // ---------- sigma MLP layer 1: 64 -> 64, relu (1-tile B lookahead pipeline) ----------
    {
      v16h a0 = load_a(actA, lane, 0);
      v16h a1 = load_a(actA, lane, 1);
      v16h b0 = load_b(sm + OFF_W1, 4, 0, 0, lane);
      v16h b1 = load_b(sm + OFF_W1, 4, 1, 0, lane);
      #pragma unroll
      for (int nt = 0; nt < 4; ++nt) {
        v8f acc = ZACC;
        acc = wmma16(a0, b0, acc);
        acc = wmma16(a1, b1, acc);
        if (nt < 3) {  // issue next tile's B loads before consuming acc (fills hazard slots)
          b0 = load_b(sm + OFF_W1, 4, 0, nt + 1, lane);
          b1 = load_b(sm + OFF_W1, 4, 1, nt + 1, lane);
        }
        v8h hv = relu_h8(acc);
        #pragma unroll
        for (int r = 0; r < 8; ++r)
          actA[(mrb + r) * AS + nt * 16 + m] = hv[r];
      }
    }
    // ---------- sigma MLP layer 2: 64 -> 65 (no relu): sigma out + geo_feat -> actC ----------
    {
      v16h a0 = load_a(actA, lane, 0);
      v16h a1 = load_a(actA, lane, 1);
      v16h b0 = load_b(sm + OFF_W2, 5, 0, 0, lane);
      v16h b1 = load_b(sm + OFF_W2, 5, 1, 0, lane);
      #pragma unroll
      for (int nt = 0; nt < 5; ++nt) {
        v8f acc = ZACC;
        acc = wmma16(a0, b0, acc);
        acc = wmma16(a1, b1, acc);
        if (nt < 4) {
          b0 = load_b(sm + OFF_W2, 5, 0, nt + 1, lane);
          b1 = load_b(sm + OFF_W2, 5, 1, nt + 1, lane);
        }
        const int cidx = nt * 16 + m;
        #pragma unroll
        for (int r = 0; r < 8; ++r) {
          const float v = acc[r];
          if (cidx == 0) {
            const int ps = pbase + mrb + r;
            if (ps < npts) __builtin_nontemporal_store(v, out + ps);        // sigma
          } else if (cidx <= 64) {
            actC[(mrb + r) * AS + 16 + (cidx - 1)] = (_Float16)v;           // geo_feat
          }
        }
      }
    }
    // ---------- color MLP layer 0: 96(pad of 80) -> 64, relu ----------
    {
      v16h a0 = load_a(actC, lane, 0);
      v16h a1 = load_a(actC, lane, 1);
      v16h a2 = load_a(actC, lane, 2);
      v16h b0 = load_b(sm + OFF_WC0, 4, 0, 0, lane);
      v16h b1 = load_b(sm + OFF_WC0, 4, 1, 0, lane);
      v16h b2 = load_b(sm + OFF_WC0, 4, 2, 0, lane);
      #pragma unroll
      for (int nt = 0; nt < 4; ++nt) {
        v8f acc = ZACC;
        acc = wmma16(a0, b0, acc);
        acc = wmma16(a1, b1, acc);
        acc = wmma16(a2, b2, acc);
        if (nt < 3) {
          b0 = load_b(sm + OFF_WC0, 4, 0, nt + 1, lane);
          b1 = load_b(sm + OFF_WC0, 4, 1, nt + 1, lane);
          b2 = load_b(sm + OFF_WC0, 4, 2, nt + 1, lane);
        }
        v8h hv = relu_h8(acc);
        #pragma unroll
        for (int r = 0; r < 8; ++r)
          actA[(mrb + r) * AS + nt * 16 + m] = hv[r];
      }
    }
    // ---------- color MLP layer 1: 64 -> 3 (padded to 16), sigmoid ----------
    {
      v16h a0 = load_a(actA, lane, 0);
      v16h a1 = load_a(actA, lane, 1);
      v16h b0 = load_b(sm + OFF_WC1, 1, 0, 0, lane);
      v16h b1 = load_b(sm + OFF_WC1, 1, 1, 0, lane);
      v8f acc = ZACC;
      acc = wmma16(a0, b0, acc);
      acc = wmma16(a1, b1, acc);
      if (m < 3) {
        #pragma unroll
        for (int r = 0; r < 8; ++r) {
          const int px = pbase + mrb + r;
          if (px < npts) {
            const float sg = 1.f / (1.f + __expf(-acc[r]));
            __builtin_nontemporal_store(sg * 1.002f - 0.001f, out + npts + px * 3 + m);
          }
        }
      }
    }
  }
}

extern "C" void kernel_launch(void* const* d_in, const int* in_sizes, int n_in,
                              void* d_out, int out_size, void* d_ws, size_t ws_size,
                              hipStream_t stream) {
  const float* x     = (const float*)d_in[0];
  const float* dvec  = (const float*)d_in[1];
  const float* table = (const float*)d_in[2];
  const float* ws0   = (const float*)d_in[3];
  const float* ws1   = (const float*)d_in[4];
  const float* ws2   = (const float*)d_in[5];
  const float* wc0   = (const float*)d_in[6];
  const float* wc1   = (const float*)d_in[7];
  float* out = (float*)d_out;

  const int npts = in_sizes[0] / 3;
  const int nb   = (npts + 63) / 64;           // 64 points per block (4 waves x 16)
  const int grid = nb < 8192 ? nb : 8192;      // persistent-ish; amortize weight staging
  ngp_fused_kernel<<<dim3(grid), dim3(128), 0, stream>>>(
      x, dvec, table, ws0, ws1, ws2, wc0, wc1, out, npts, nb);
}
